// FlowGNN_50002009260670
// MI455X (gfx1250) — matrix operand order repaired and checked by reading, not compiled, not run on hardware
//
#include <hip/hip_runtime.h>
#include <hip/hip_bf16.h>

typedef __attribute__((ext_vector_type(16))) _Float16 v16h;
typedef __attribute__((ext_vector_type(8)))  float    v8f;

#define N_NODES 100000
#define DEG     32
#define BQ      2048
#define F_IN    64
#define F_H     128
#define F_OUT   64
#define ROWS1   (BQ * (DEG + 1))   // 67584 frontier rows

// Load one wave's A-fragment (16x32 fp16 tile) from a row-major fp16 LDS tile.
// 16-bit A layout (contiguous, matches B): lane 0-15 -> row M=lane, K = {0..15} in v0..7;
// lane 16-31 -> same rows, K offset +16.  koff = kstep*32 + (lane<16 ? 0 : 16).
__device__ __forceinline__ v16h load_a_frag(const _Float16* lds_row, int koff) {
  union { v16h v; unsigned u[8]; } r;
  const unsigned* p = (const unsigned*)lds_row;
#pragma unroll
  for (int vg = 0; vg < 8; ++vg) {
    int f16off = 2 * vg;
    r.u[vg] = p[(koff + f16off) >> 1];
  }
  return r.v;
}

// ---------------------------------------------------------------------------
// Kernel 1: per-frontier-node masked mean over 32 temporal neighbors (+self),
// then h1 = relu(mean @ W0 + b0) via WMMA; h1 pre-multiplied by mask_full.
// One workgroup = 16 frontier rows; 8 waves each own one 16x16 N-tile of F_H=128.
// ---------------------------------------------------------------------------
__global__ __launch_bounds__(256) void gnn_layer1(
    const float* __restrict__ x, const float* __restrict__ times,
    const float* __restrict__ ts, const float* __restrict__ W0,
    const float* __restrict__ b0, const int* __restrict__ batch,
    const int* __restrict__ idx, _Float16* __restrict__ h1)
{
  __shared__ __align__(16) _Float16 Wt[F_H * F_IN];    // W0^T as [n][k] fp16, 16KB
  __shared__ __align__(16) _Float16 Atile[16 * 72];    // 16 rows x 64 fp16, padded
  __shared__ int   nbr[16][DEG];                       // masked neighbor ids (-1 = masked out)
  __shared__ int   nodeArr[16];
  __shared__ float tqArr[16];
  __shared__ float rowScale[16];                       // mask_full as 0/1
  __shared__ int   cnt[16];

  const int tid = threadIdx.x;
  const int wg  = blockIdx.x;

  // Stage W0 (row-major [K=64][N=128]) into LDS transposed fp16 [N][K].
  for (int i = tid; i < F_IN * F_H; i += 256) {
    int k = i >> 7, n = i & 127;
    Wt[n * F_IN + k] = (_Float16)W0[i];
  }

  // Row metadata: frontier node id, query time, mask_full, init count (self).
  if (tid < 16) {
    int m = wg * 16 + tid;
    int b = m / 33, j = m - b * 33;
    int bn  = batch[b];
    float tb = times[b];
    int node; float tq; float sc;
    if (j < DEG) {
      int e = bn * DEG + j;
      node = idx[e];
      tq   = ts[e];
      sc   = (tq <= tb) ? 1.0f : 0.0f;
    } else {
      node = bn; tq = tb; sc = 1.0f;   // self row
    }
    nodeArr[tid] = node; tqArr[tid] = tq; rowScale[tid] = sc; cnt[tid] = 1;
  }
  __syncthreads();

  // Stage masked neighbor ids once (avoids 16x redundant edge loads).
  for (int e = tid; e < 16 * DEG; e += 256) {
    int r = e >> 5, d = e & 31;
    int ee = nodeArr[r] * DEG + d;
    int nb = idx[ee];
    bool ok = (ts[ee] <= tqArr[r]);
    nbr[r][d] = ok ? nb : -1;
    if (ok) atomicAdd(&cnt[r], 1);
  }
  __syncthreads();

  // Masked mean: 16 threads x float4 per row; 256B fully-coalesced row gathers
  // (x is 25.6MB -> L2 resident on the 192MB L2).
  {
    int r = tid >> 4, f0 = (tid & 15) * 4;
    float4 acc = *(const float4*)(x + (size_t)nodeArr[r] * F_IN + f0);  // self
#pragma unroll 4
    for (int d = 0; d < DEG; ++d) {
      int nb = nbr[r][d];
      if (nb >= 0) {
        float4 v = *(const float4*)(x + (size_t)nb * F_IN + f0);
        acc.x += v.x; acc.y += v.y; acc.z += v.z; acc.w += v.w;
      }
    }
    float s = 1.0f / (float)cnt[r];
    _Float16* a = &Atile[r * 72 + f0];
    a[0] = (_Float16)(acc.x * s); a[1] = (_Float16)(acc.y * s);
    a[2] = (_Float16)(acc.z * s); a[3] = (_Float16)(acc.w * s);
  }
  __syncthreads();

  // GEMM [16 x 64] @ [64 x 128]: wave w -> N-tile w*16, K-loop of 2.
  {
    int wv = tid >> 5, lane = tid & 31;
    int nl    = wv * 16 + (lane & 15);
    int ahalf = (lane < 16) ? 0 : 16;
    int bhalf = (lane < 16) ? 0 : 16;  // B layout: lanes 16-31 hold K+16
    const _Float16* arow = &Atile[(lane & 15) * 72];

    v8f c = {};
#pragma unroll
    for (int s = 0; s < 2; ++s) {
      v16h a  = load_a_frag(arow, s * 32 + ahalf);
      v16h bb = *(const v16h*)(&Wt[(size_t)nl * F_IN + s * 32 + bhalf]);
      c = __builtin_amdgcn_wmma_f32_16x16x32_f16(false, a, false, bb,
                                                 (short)0, c, false, false);
    }
    float bias = b0[nl];
    int rbase = (lane < 16) ? 0 : 8;   // C layout: lanes 16-31 hold M+8
#pragma unroll
    for (int i = 0; i < 8; ++i) {
      int r = rbase + i;
      float v = c[i] + bias;
      v = v > 0.0f ? v : 0.0f;
      v *= rowScale[r];                // fold mask_full into stored h1
      h1[(size_t)(wg * 16 + r) * F_H + nl] = (_Float16)v;
    }
  }
}

// ---------------------------------------------------------------------------
// Kernel 2: agg1 = sum_j h1_masked / cnt1 ; out = relu(agg1 @ W1 + b1).
// One workgroup = 16 query rows; 4 waves each own one 16x16 N-tile of F_OUT=64.
// ---------------------------------------------------------------------------
__global__ __launch_bounds__(128) void gnn_layer2(
    const _Float16* __restrict__ h1, const float* __restrict__ times,
    const float* __restrict__ ts, const float* __restrict__ W1,
    const float* __restrict__ b1, const int* __restrict__ batch,
    float* __restrict__ out)
{
  __shared__ __align__(16) _Float16 Wt[F_OUT * F_H];   // W1^T as [n][k] fp16, 16KB
  __shared__ __align__(16) _Float16 Atile[16 * 136];   // 16 rows x 128 fp16, padded
  __shared__ int cnt[16];

  const int tid = threadIdx.x;
  const int wg  = blockIdx.x;

  // Stage W1 (row-major [K=128][N=64]) into LDS transposed fp16 [N][K].
  for (int i = tid; i < F_H * F_OUT; i += 128) {
    int k = i >> 6, n = i & 63;
    Wt[n * F_H + k] = (_Float16)W1[i];
  }
  if (tid < 16) cnt[tid] = 1;          // self
  __syncthreads();

  // cnt1 = 1 + popcount(mask1): 8 threads x 4 edges per row.
  {
    int r = tid >> 3, q = tid & 7;
    int b  = wg * 16 + r;
    int bn = batch[b];
    float tb = times[b];
    int local = 0;
#pragma unroll
    for (int d = q * 4; d < q * 4 + 4; ++d)
      local += (ts[bn * DEG + d] <= tb) ? 1 : 0;
    if (local) atomicAdd(&cnt[r], local);
  }
  __syncthreads();

  // Sum 33 masked h1 rows (fp16 in, fp32 accumulate), scale, fp16 A-tile.
  {
    int r = tid >> 3, c0 = (tid & 7) * 16;
    int b = wg * 16 + r;
    float acc[16];
#pragma unroll
    for (int i = 0; i < 16; ++i) acc[i] = 0.0f;
    const _Float16* base = h1 + ((size_t)b * 33) * F_H + c0;
    for (int j = 0; j < 33; ++j) {
      v16h h = *(const v16h*)(base + (size_t)j * F_H);
#pragma unroll
      for (int i = 0; i < 16; ++i) acc[i] += (float)h[i];
    }
    float s = 1.0f / (float)cnt[r];
    _Float16* a = &Atile[r * 136 + c0];
#pragma unroll
    for (int i = 0; i < 16; ++i) a[i] = (_Float16)(acc[i] * s);
  }
  __syncthreads();

  // GEMM [16 x 128] @ [128 x 64]: wave w -> N-tile w*16, K-loop of 4.
  {
    int wv = tid >> 5, lane = tid & 31;
    int nl    = wv * 16 + (lane & 15);
    int ahalf = (lane < 16) ? 0 : 16;
    int bhalf = (lane < 16) ? 0 : 16;
    const _Float16* arow = &Atile[(lane & 15) * 136];

    v8f c = {};
#pragma unroll
    for (int s = 0; s < 4; ++s) {
      v16h a  = load_a_frag(arow, s * 32 + ahalf);
      v16h bb = *(const v16h*)(&Wt[(size_t)nl * F_H + s * 32 + bhalf]);
      c = __builtin_amdgcn_wmma_f32_16x16x32_f16(false, a, false, bb,
                                                 (short)0, c, false, false);
    }
    float bias = b1[nl];
    int rbase = (lane < 16) ? 0 : 8;
#pragma unroll
    for (int i = 0; i < 8; ++i) {
      float v = c[i] + bias;
      v = v > 0.0f ? v : 0.0f;
      out[(size_t)(wg * 16 + rbase + i) * F_OUT + nl] = v;
    }
  }
}

extern "C" void kernel_launch(void* const* d_in, const int* in_sizes, int n_in,
                              void* d_out, int out_size, void* d_ws, size_t ws_size,
                              hipStream_t stream) {
  const float* x     = (const float*)d_in[0];
  const float* times = (const float*)d_in[1];
  const float* ts    = (const float*)d_in[2];
  const float* W0    = (const float*)d_in[3];
  const float* b0    = (const float*)d_in[4];
  const float* W1    = (const float*)d_in[5];
  const float* b1    = (const float*)d_in[6];
  const int*   batch = (const int*)d_in[7];
  const int*   idx   = (const int*)d_in[8];

  _Float16* h1 = (_Float16*)d_ws;     // 67584 x 128 fp16 = 17.3 MB scratch
  float*    out = (float*)d_out;

  gnn_layer1<<<ROWS1 / 16, 256, 0, stream>>>(x, times, ts, W0, b0, batch, idx, h1);
  gnn_layer2<<<BQ / 16, 128, 0, stream>>>(h1, times, ts, W1, b1, batch, out);

  (void)in_sizes; (void)n_in; (void)out_size; (void)ws_size;
}
